// Semantic_Fusion_Network_41094247088734
// MI455X (gfx1250) — matrix operand order repaired and checked by reading, not compiled, not run on hardware
//
#include <hip/hip_runtime.h>
#include <cstddef>

// ---------------- types ----------------
typedef __attribute__((ext_vector_type(16))) __bf16        v16bf;
typedef __attribute__((ext_vector_type(8)))  float         v8f;
typedef __attribute__((ext_vector_type(4)))  unsigned int  v4u;
typedef __attribute__((ext_vector_type(2)))  unsigned int  v2u;

union Frag {
    v16bf v;
    v4u   q[2];
};

#define BB      8
#define HW      64
#define NC      256       // NORM_NC
#define SLN     512
#define LNC     19
#define NPIX    4096      // 64*64
#define NPIXALL 32768     // 8*64*64
#define KPAIR   12800     // 25*512
#define KSH     800       // 25*32 (19 padded to 32)
#define GOFF512 (NPIXALL * SLN)   // guard offset (halfs) in mid/act buffers
#define GOFFSEG (NPIXALL * 32)    // guard offset (halfs) in seg buffer

__device__ __forceinline__ unsigned short f2bf(float f) {
    unsigned int u = __float_as_uint(f);
    u += 0x7fffu + ((u >> 16) & 1u);   // round-to-nearest-even
    return (unsigned short)(u >> 16);
}

__device__ __forceinline__ v8f wmma_bf16(const Frag& a, const Frag& b, v8f c) {
    return __builtin_amdgcn_wmma_f32_16x16x32_bf16(false, a.v, false, b.v, (short)0, c, false, false);
}

// ---------------- 0) zero the guard tails (512 bf16 each) ----------------
__global__ void guard_init_kernel(unsigned int* __restrict__ g0,
                                  unsigned int* __restrict__ g1,
                                  unsigned int* __restrict__ g2) {
    g0[threadIdx.x] = 0u;    // 256 threads * 4B = 512 bf16 zeros
    g1[threadIdx.x] = 0u;
    g2[threadIdx.x] = 0u;
}

// ---------------- 1) instance-norm stats: one block per (b,c) ----------------
__global__ __launch_bounds__(256) void stats_kernel(const float* __restrict__ x,
                                                    float* __restrict__ mean,
                                                    float* __restrict__ rstd) {
    __shared__ float ssum[256];
    __shared__ float ssq[256];
    const int bc  = blockIdx.x;
    const int tid = threadIdx.x;
    const float* p = x + (size_t)bc * NPIX;
    float s = 0.f, q = 0.f;
    for (int i = tid; i < NPIX; i += 256) { float v = p[i]; s += v; q += v * v; }
    ssum[tid] = s; ssq[tid] = q;
    __syncthreads();
    for (int st = 128; st > 0; st >>= 1) {
        if (tid < st) { ssum[tid] += ssum[tid + st]; ssq[tid] += ssq[tid + st]; }
        __syncthreads();
    }
    if (tid == 0) {
        float m = ssum[0] * (1.f / NPIX);
        float v = ssq[0] * (1.f / NPIX) - m * m;
        mean[bc] = m;
        rstd[bc] = rsqrtf(v + 1e-5f);
    }
}

// ---------------- 2) mu[b,j,e] = relu(sum_d style[b,j,d]*fc_w[j,e,d] + fc_b[j,e]) ----------------
__global__ __launch_bounds__(256) void mu_kernel(const float* __restrict__ style,
                                                 const float* __restrict__ fcw,
                                                 const float* __restrict__ fcb,
                                                 float* __restrict__ mu) {
    int t = blockIdx.x * 256 + threadIdx.x;
    if (t >= BB * LNC * SLN) return;
    int e = t & 511;
    int j = (t >> 9) % LNC;
    int b = t / (LNC * SLN);
    const float4* srow = (const float4*)(style + (size_t)(b * LNC + j) * SLN);
    const float4* wrow = (const float4*)(fcw + (size_t)(j * SLN + e) * SLN);
    float acc = 0.f;
    for (int i = 0; i < SLN / 4; ++i) {
        float4 a = srow[i], w = wrow[i];
        acc += a.x * w.x + a.y * w.y + a.z * w.z + a.w * w.w;
    }
    acc += fcb[j * SLN + e];
    mu[t] = acc > 0.f ? acc : 0.f;
}

// ---------------- 3) segmap (B,19,H,W) f32 -> NHWC bf16 padded to 32 ch ----------------
__global__ __launch_bounds__(256) void seg_pack_kernel(const float* __restrict__ seg,
                                                       unsigned short* __restrict__ dst) {
    int t = blockIdx.x * 256 + threadIdx.x;           // NPIXALL*32
    int c = t & 31;
    int p = t >> 5;
    int b = p >> 12;
    int yx = p & (NPIX - 1);
    float v = (c < LNC) ? seg[(size_t)(b * LNC + c) * NPIX + yx] : 0.f;
    dst[t] = f2bf(v);
}

// ---------------- 4) weight pack: (O=256,I=512,5,5)x2 -> [512][12800] bf16, blend-scaled ----------------
__global__ __launch_bounds__(256) void pack_pair_kernel(const float* __restrict__ wg,
                                                        const float* __restrict__ wb,
                                                        const float* __restrict__ blg,
                                                        const float* __restrict__ blb,
                                                        int invert,
                                                        unsigned short* __restrict__ dst) {
    int t = blockIdx.x * 256 + threadIdx.x;           // 512*12800
    int n = t / KPAIR;
    int k = t - n * KPAIR;
    int tap = k >> 9;
    int c   = k & 511;
    int o   = n & 255;
    float sc;
    const float* src;
    if (n < 256) {
        float g = 1.f / (1.f + __expf(-blg[0]));
        sc = invert ? (1.f - g) : g;
        src = wg;
    } else {
        float g = 1.f / (1.f + __expf(-blb[0]));
        sc = invert ? (1.f - g) : g;
        src = wb;
    }
    float v = src[(size_t)(o * SLN + c) * 25 + tap] * sc;
    dst[t] = f2bf(v);
}

// ---------------- 5) shared-conv weight pack: (512,19,5,5) -> [512][800] bf16 ----------------
__global__ __launch_bounds__(256) void pack_shared_kernel(const float* __restrict__ sw,
                                                          unsigned short* __restrict__ dst) {
    int t = blockIdx.x * 256 + threadIdx.x;           // 512*800
    if (t >= SLN * KSH) return;
    int n = t / KSH;
    int k = t - n * KSH;
    int tap = k >> 5;
    int c   = k & 31;
    float v = (c < LNC) ? sw[(size_t)(n * LNC + c) * 25 + tap] : 0.f;
    dst[t] = f2bf(v);
}

// ---------------- 6) middle_avg gather: per pixel last label -> mu row, NHWC bf16 ----------------
__global__ __launch_bounds__(256) void midsel_kernel(const float* __restrict__ seg,
                                                     const float* __restrict__ mu,
                                                     unsigned short* __restrict__ mid) {
    int t = blockIdx.x * 256 + threadIdx.x;           // NPIXALL * 128 (4 ch each)
    int c0 = (t & 127) << 2;
    int p  = t >> 7;
    int b  = p >> 12;
    int yx = p & (NPIX - 1);
    int j = -1;
    for (int jj = LNC - 1; jj >= 0; --jj) {
        if (seg[(size_t)(b * LNC + jj) * NPIX + yx] > 0.f) { j = jj; break; }
    }
    v2u o2 = v2u{0u, 0u};
    if (j >= 0) {
        float4 vv = *(const float4*)(mu + (size_t)(b * LNC + j) * SLN + c0);
        o2.x = (unsigned int)f2bf(vv.x) | ((unsigned int)f2bf(vv.y) << 16);
        o2.y = (unsigned int)f2bf(vv.z) | ((unsigned int)f2bf(vv.w) << 16);
    }
    *(v2u*)(mid + (size_t)p * SLN + c0) = o2;
}

// ---------------- wave tile: M32 x N64 per wave; 8 waves of a block share n0 ----------------
struct Tile {
    int b, y, x0, n0, lrow, hi;
};
__device__ __forceinline__ Tile decode_tile() {
    Tile tl;
    const int lane = threadIdx.x & 31;
    const int wave = threadIdx.x >> 5;
    const int mt   = (blockIdx.x >> 3) * 8 + wave;    // [0,1024): M32 tiles
    tl.n0 = (blockIdx.x & 7) * 64;
    tl.b = mt >> 7;
    tl.y = (mt >> 1) & 63;
    tl.x0 = (mt & 1) << 5;
    tl.lrow = lane & 15;
    tl.hi = lane >> 4;
    return tl;
}

// accumulate one conv phase: input NHWC (512 ch, zero-guard at GOFF512), weights [512][KPAIR]
__device__ __forceinline__ void gemm_phase512(const unsigned short* __restrict__ in,
                                              const unsigned short* __restrict__ W,
                                              const Tile& tl, v8f (&acc)[2][4]) {
    const unsigned short* wbase = W + (size_t)(tl.n0 + tl.lrow) * KPAIR + tl.hi * 16;
#pragma unroll 1
    for (int tap = 0; tap < 25; ++tap) {
        const int ky = tap / 5, kx = tap - ky * 5;
        const int yy = tl.y + ky - 2;
        const bool rowok = (yy >= 0) & (yy < HW);
        const int xx0 = tl.x0 + tl.lrow + kx - 2;
        const int xx1 = xx0 + 16;
        // branchless offset select into the SAME buffer (guard tail holds zeros)
        const int po0 = (rowok & (xx0 >= 0) & (xx0 < HW)) ? ((tl.b * HW + yy) * HW + xx0) * SLN : GOFF512;
        const int po1 = (rowok & (xx1 >= 0) & (xx1 < HW)) ? ((tl.b * HW + yy) * HW + xx1) * SLN : GOFF512;
        const unsigned short* ap0 = in + po0 + tl.hi * 16;
        const unsigned short* ap1 = in + po1 + tl.hi * 16;
        const unsigned short* wp  = wbase + tap * SLN;
#pragma unroll 1
        for (int cg = 0; cg < SLN; cg += 32) {
            Frag a0, a1, b0, b1, b2, b3;
            a0.q[0] = *(const v4u*)(ap0);
            a0.q[1] = *(const v4u*)(ap0 + 8);
            a1.q[0] = *(const v4u*)(ap1);
            a1.q[1] = *(const v4u*)(ap1 + 8);
            b0.q[0] = *(const v4u*)(wp);
            b0.q[1] = *(const v4u*)(wp + 8);
            b1.q[0] = *(const v4u*)(wp + 16 * KPAIR);
            b1.q[1] = *(const v4u*)(wp + 16 * KPAIR + 8);
            b2.q[0] = *(const v4u*)(wp + 32 * KPAIR);
            b2.q[1] = *(const v4u*)(wp + 32 * KPAIR + 8);
            b3.q[0] = *(const v4u*)(wp + 48 * KPAIR);
            b3.q[1] = *(const v4u*)(wp + 48 * KPAIR + 8);
            acc[0][0] = wmma_bf16(a0, b0, acc[0][0]);
            acc[1][0] = wmma_bf16(a1, b0, acc[1][0]);
            acc[0][1] = wmma_bf16(a0, b1, acc[0][1]);
            acc[1][1] = wmma_bf16(a1, b1, acc[1][1]);
            acc[0][2] = wmma_bf16(a0, b2, acc[0][2]);
            acc[1][2] = wmma_bf16(a1, b2, acc[1][2]);
            acc[0][3] = wmma_bf16(a0, b3, acc[0][3]);
            acc[1][3] = wmma_bf16(a1, b3, acc[1][3]);
            ap0 += 32; ap1 += 32; wp += 32;
        }
    }
}

// ---------------- 7) shared conv: seg_nhwc(32ch) x Ws[512][800] -> relu -> actv NHWC bf16 ----------------
__global__ __launch_bounds__(256) void conv_shared_kernel(const unsigned short* __restrict__ seg,
                                                          const unsigned short* __restrict__ Ws,
                                                          const float* __restrict__ ssb,
                                                          unsigned short* __restrict__ actv) {
    const Tile tl = decode_tile();

    v8f acc[2][4];
#pragma unroll
    for (int mi = 0; mi < 2; ++mi)
#pragma unroll
        for (int t = 0; t < 4; ++t) acc[mi][t] = v8f{};

    const unsigned short* wbase = Ws + (size_t)(tl.n0 + tl.lrow) * KSH + tl.hi * 16;
#pragma unroll 1
    for (int tap = 0; tap < 25; ++tap) {
        const int ky = tap / 5, kx = tap - ky * 5;
        const int yy = tl.y + ky - 2;
        const bool rowok = (yy >= 0) & (yy < HW);
        const int xx0 = tl.x0 + tl.lrow + kx - 2;
        const int xx1 = xx0 + 16;
        const int po0 = (rowok & (xx0 >= 0) & (xx0 < HW)) ? ((tl.b * HW + yy) * HW + xx0) * 32 : GOFFSEG;
        const int po1 = (rowok & (xx1 >= 0) & (xx1 < HW)) ? ((tl.b * HW + yy) * HW + xx1) * 32 : GOFFSEG;
        const unsigned short* ap0 = seg + po0 + tl.hi * 16;
        const unsigned short* ap1 = seg + po1 + tl.hi * 16;
        const unsigned short* wp  = wbase + tap * 32;
        Frag a0, a1, b0, b1, b2, b3;
        a0.q[0] = *(const v4u*)(ap0);
        a0.q[1] = *(const v4u*)(ap0 + 8);
        a1.q[0] = *(const v4u*)(ap1);
        a1.q[1] = *(const v4u*)(ap1 + 8);
        b0.q[0] = *(const v4u*)(wp);
        b0.q[1] = *(const v4u*)(wp + 8);
        b1.q[0] = *(const v4u*)(wp + 16 * KSH);
        b1.q[1] = *(const v4u*)(wp + 16 * KSH + 8);
        b2.q[0] = *(const v4u*)(wp + 32 * KSH);
        b2.q[1] = *(const v4u*)(wp + 32 * KSH + 8);
        b3.q[0] = *(const v4u*)(wp + 48 * KSH);
        b3.q[1] = *(const v4u*)(wp + 48 * KSH + 8);
        acc[0][0] = wmma_bf16(a0, b0, acc[0][0]);
        acc[1][0] = wmma_bf16(a1, b0, acc[1][0]);
        acc[0][1] = wmma_bf16(a0, b1, acc[0][1]);
        acc[1][1] = wmma_bf16(a1, b1, acc[1][1]);
        acc[0][2] = wmma_bf16(a0, b2, acc[0][2]);
        acc[1][2] = wmma_bf16(a1, b2, acc[1][2]);
        acc[0][3] = wmma_bf16(a0, b3, acc[0][3]);
        acc[1][3] = wmma_bf16(a1, b3, acc[1][3]);
    }

#pragma unroll
    for (int mi = 0; mi < 2; ++mi) {
        const int pixbase = (tl.b * HW + tl.y) * HW + tl.x0 + mi * 16;
#pragma unroll
        for (int t = 0; t < 4; ++t) {
            const int n = tl.n0 + t * 16 + tl.lrow;
            const float bias = ssb[n];
#pragma unroll
            for (int r = 0; r < 8; ++r) {
                const int m = r + tl.hi * 8;
                float v = acc[mi][t][r] + bias;
                v = v > 0.f ? v : 0.f;
                actv[(size_t)(pixbase + m) * SLN + n] = f2bf(v);
            }
        }
    }
}

// ---------------- 8) fused final conv: 4 convs + sigmoid blend in one accumulator ----------------
__global__ __launch_bounds__(256) void conv_final_kernel(const unsigned short* __restrict__ mid,
                                                         const unsigned short* __restrict__ actv,
                                                         const unsigned short* __restrict__ wavg,
                                                         const unsigned short* __restrict__ wsp,
                                                         const float* __restrict__ cgb,
                                                         const float* __restrict__ cbb,
                                                         const float* __restrict__ sgb,
                                                         const float* __restrict__ sbb,
                                                         const float* __restrict__ blg,
                                                         const float* __restrict__ blb,
                                                         float* __restrict__ gf,
                                                         float* __restrict__ bfo) {
    const Tile tl = decode_tile();

    v8f acc[2][4];
#pragma unroll
    for (int mi = 0; mi < 2; ++mi)
#pragma unroll
        for (int t = 0; t < 4; ++t) acc[mi][t] = v8f{};

    gemm_phase512(mid, wavg, tl, acc);
    gemm_phase512(actv, wsp, tl, acc);

    const float ga = 1.f / (1.f + __expf(-blg[0]));
    const float ba = 1.f / (1.f + __expf(-blb[0]));
#pragma unroll
    for (int t = 0; t < 4; ++t) {
        const int n = tl.n0 + t * 16 + tl.lrow;
        float bias;
        float* dst;
        if (n < NC) {
            bias = ga * cgb[n] + (1.f - ga) * sgb[n];
            dst  = gf + (size_t)(tl.b * NC + n) * NPIX;
        } else {
            const int nn = n - NC;
            bias = ba * cbb[nn] + (1.f - ba) * sbb[nn];
            dst  = bfo + (size_t)(tl.b * NC + nn) * NPIX;
        }
#pragma unroll
        for (int mi = 0; mi < 2; ++mi) {
            const int yxbase = tl.y * HW + tl.x0 + mi * 16;
#pragma unroll
            for (int r = 0; r < 8; ++r) {
                const int m = r + tl.hi * 8;
                dst[yxbase + m] = acc[mi][t][r] + bias;
            }
        }
    }
}

// ---------------- 9) final: out = norm*(1+gamma_final) + beta_final ----------------
__global__ __launch_bounds__(256) void final_kernel(const float* __restrict__ x,
                                                    const float* __restrict__ mean,
                                                    const float* __restrict__ rstd,
                                                    const float* __restrict__ gf,
                                                    const float* __restrict__ bfo,
                                                    float* __restrict__ out) {
    int t = blockIdx.x * 256 + threadIdx.x;           // 8*256*4096
    int bc = t >> 12;
    float nm = (x[t] - mean[bc]) * rstd[bc];
    out[t] = nm * (1.f + gf[t]) + bfo[t];
}

// ---------------- launcher ----------------
extern "C" void kernel_launch(void* const* d_in, const int* in_sizes, int n_in,
                              void* d_out, int out_size, void* d_ws, size_t ws_size,
                              hipStream_t stream) {
    (void)in_sizes; (void)n_in; (void)out_size; (void)ws_size;
    const float* x      = (const float*)d_in[0];
    const float* segmap = (const float*)d_in[1];
    const float* style  = (const float*)d_in[2];
    const float* fcw    = (const float*)d_in[3];
    const float* fcb    = (const float*)d_in[4];
    const float* cgw    = (const float*)d_in[5];
    const float* cgb    = (const float*)d_in[6];
    const float* cbw    = (const float*)d_in[7];
    const float* cbb    = (const float*)d_in[8];
    const float* ssw    = (const float*)d_in[9];
    const float* ssb    = (const float*)d_in[10];
    const float* sgw    = (const float*)d_in[11];
    const float* sgb    = (const float*)d_in[12];
    const float* sbw    = (const float*)d_in[13];
    const float* sbb    = (const float*)d_in[14];
    const float* blg    = (const float*)d_in[15];
    const float* blb    = (const float*)d_in[16];
    float* out = (float*)d_out;

    char* p = (char*)d_ws;
    auto alloc = [&](size_t bytes) -> char* {
        char* r = p;
        p += (bytes + 255) & ~(size_t)255;
        return r;
    };
    float* mean          = (float*)alloc((size_t)BB * NC * 4);
    float* rstd          = (float*)alloc((size_t)BB * NC * 4);
    float* mu            = (float*)alloc((size_t)BB * LNC * SLN * 4);
    unsigned short* seg  = (unsigned short*)alloc(((size_t)NPIXALL * 32 + 512) * 2);
    unsigned short* Ws   = (unsigned short*)alloc((size_t)SLN * KSH * 2);
    unsigned short* mid  = (unsigned short*)alloc(((size_t)NPIXALL * SLN + 512) * 2);
    unsigned short* act  = (unsigned short*)alloc(((size_t)NPIXALL * SLN + 512) * 2);
    unsigned short* Wav  = (unsigned short*)alloc((size_t)SLN * KPAIR * 2);
    unsigned short* Wsp  = (unsigned short*)alloc((size_t)SLN * KPAIR * 2);
    float* gf            = (float*)alloc((size_t)BB * NC * NPIX * 4);
    float* bf            = (float*)alloc((size_t)BB * NC * NPIX * 4);

    guard_init_kernel<<<1, 256, 0, stream>>>((unsigned int*)(seg + GOFFSEG),
                                             (unsigned int*)(mid + GOFF512),
                                             (unsigned int*)(act + GOFF512));
    stats_kernel<<<BB * NC, 256, 0, stream>>>(x, mean, rstd);
    mu_kernel<<<(BB * LNC * SLN + 255) / 256, 256, 0, stream>>>(style, fcw, fcb, mu);
    seg_pack_kernel<<<(NPIXALL * 32) / 256, 256, 0, stream>>>(segmap, seg);
    pack_shared_kernel<<<(SLN * KSH + 255) / 256, 256, 0, stream>>>(ssw, Ws);
    pack_pair_kernel<<<(SLN * KPAIR) / 256, 256, 0, stream>>>(cgw, cbw, blg, blb, 0, Wav);
    pack_pair_kernel<<<(SLN * KPAIR) / 256, 256, 0, stream>>>(sgw, sbw, blg, blb, 1, Wsp);
    midsel_kernel<<<(NPIXALL * 128) / 256, 256, 0, stream>>>(segmap, mu, mid);
    conv_shared_kernel<<<1024, 256, 0, stream>>>(seg, Ws, ssb, act);
    conv_final_kernel<<<1024, 256, 0, stream>>>(mid, act, Wav, Wsp, cgb, cbb, sgb, sbb, blg, blb, gf, bf);
    final_kernel<<<(BB * NC * NPIX) / 256, 256, 0, stream>>>(x, mean, rstd, gf, bf, out);
}
